// DenoisingNetwork_14250701488264
// MI455X (gfx1250) — matrix operand order repaired and checked by reading, not compiled, not run on hardware
//
#include <hip/hip_runtime.h>

// ---------- types ----------
typedef __bf16   bf16;
typedef __attribute__((ext_vector_type(16))) __bf16 v16bf;
typedef __attribute__((ext_vector_type(8)))  float  v8f;
typedef __attribute__((ext_vector_type(4)))  unsigned u32x4;
typedef __attribute__((ext_vector_type(4)))  float  f32x4;
typedef __attribute__((ext_vector_type(2)))  float  f32x2;

union Frag { v16bf v; u32x4 u[2]; };

#define Nn 2048
#define Ee 65536
#define Hh 256
#define LL 5
#define KK 20
#define ET 4

// ---------- weight prep: fp32 [K,N] -> bf16 [N,K] (per layer), or plain convert ----------
__global__ void k_transcvt(const float* __restrict__ in, bf16* __restrict__ out,
                           int K, int N, int total) {
  int kn = K * N;
  for (int idx = blockIdx.x * blockDim.x + threadIdx.x; idx < total;
       idx += gridDim.x * blockDim.x) {
    int l = idx / kn, r = idx - l * kn;
    int n = r / K, k = r - n * K;
    out[idx] = (bf16)in[(size_t)l * kn + (size_t)k * N + n];
  }
}

__global__ void k_convert(const float* __restrict__ in, bf16* __restrict__ out, int total) {
  for (int idx = blockIdx.x * blockDim.x + threadIdx.x; idx < total;
       idx += gridDim.x * blockDim.x)
    out[idx] = (bf16)in[idx];
}

// ---------- h_v = x.reshape(N,28) @ node_W + node_b ----------
__global__ void k_node_proj(const float* __restrict__ x, const float* __restrict__ W,
                            const float* __restrict__ b, float* __restrict__ hv,
                            bf16* __restrict__ hvbf) {
  __shared__ float xr[28];
  int n = blockIdx.x, h = threadIdx.x;
  if (h < 28) xr[h] = x[n * 28 + h];
  __syncthreads();
  float acc = b[h];
  for (int k = 0; k < 28; k++) acc += xr[k] * W[k * Hh + h];
  hv[(size_t)n * Hh + h] = acc;
  hvbf[(size_t)n * Hh + h] = (bf16)acc;
}

// ---------- generic bf16 WMMA GEMM: C[M,N] = A[M,K] @ Bt[N,K]^T + bias (opt relu) ----------
__global__ __launch_bounds__(256) void k_gemm_bf16(
    const bf16* __restrict__ A, const bf16* __restrict__ Bt,
    const float* __restrict__ bias, float* __restrict__ C,
    int M, int N, int K, int relu) {
  int wave = threadIdx.x >> 5, lane = threadIdx.x & 31;
  int tile = blockIdx.x * 8 + wave;
  int tn = tile % (N >> 4), tm = tile / (N >> 4);
  int hi = lane >> 4, lo = lane & 15;
  const bf16* arow = A + (size_t)(tm * 16 + lo) * K + hi * 8;   // A elems 0..7 @k, 8..15 @k+16
  const bf16* brow = Bt + (size_t)(tn * 16 + lo) * K + hi * 8;  // B elems 0..7 @k, 8..15 @k+16
  v8f acc = {};
  for (int kb = 0; kb < K; kb += 32) {
    Frag a, b;
    a.u[0] = *(const u32x4*)(arow + kb);
    a.u[1] = *(const u32x4*)(arow + kb + 16);
    b.u[0] = *(const u32x4*)(brow + kb);
    b.u[1] = *(const u32x4*)(brow + kb + 16);
    acc = __builtin_amdgcn_wmma_f32_16x16x32_bf16(false, a.v, false, b.v, (short)0, acc,
                                                  false, false);
  }
  int col = tn * 16 + lo;
  float bv = bias ? bias[col] : 0.f;
  for (int i = 0; i < 8; i++) {
    int row = tm * 16 + i + hi * 8;
    float v = acc[i] + bv;
    if (relu) v = fmaxf(v, 0.f);
    C[(size_t)row * N + col] = v;
  }
}

// ---------- fused edge MLP + segment-sum: 16 edges per block, 16 waves ----------
#define CAT_S 776
#define HID_S 264
__global__ __launch_bounds__(512) void k_edge_mlp(
    const bf16* __restrict__ hvbf, const int* __restrict__ eidx,
    const float* __restrict__ eattr, const float* __restrict__ eW,
    const float* __restrict__ eB,
    const bf16* __restrict__ W1f, const float* __restrict__ b1f,
    const bf16* __restrict__ W2f, const float* __restrict__ b2f,
    const bf16* __restrict__ W1g, const float* __restrict__ b1g,
    const bf16* __restrict__ W2g, const float* __restrict__ b2g,
    float* __restrict__ agg) {
  __shared__ alignas(16) bf16 cat[16][CAT_S];   // [edge][3H] padded
  __shared__ alignas(16) bf16 hid[16][HID_S];   // relu hidden, padded
  __shared__ float mbuf[16][HID_S];             // m path result
  __shared__ int dstL[16];
  int tid = threadIdx.x;
  int e0 = blockIdx.x * 16;
  if (tid < 16) dstL[tid] = eidx[Ee + e0 + tid];  // edge_index[1] = dst
  __syncthreads();

  // ---- stage cat = [h_v[dst] | h_v[src] | h_e] as bf16 ----
  {
    int r = tid >> 5, s = tid & 31;
    int e = e0 + r;
    int d = dstL[r];
    int sr = eidx[e];                              // edge_index[0] = src
    ((u32x4*)&cat[r][0])[s]   = ((const u32x4*)(hvbf + (size_t)d  * Hh))[s];
    ((u32x4*)&cat[r][256])[s] = ((const u32x4*)(hvbf + (size_t)sr * Hh))[s];
    float ea = eattr[e];
    for (int i = 0; i < 8; i++) {
      int c = s * 8 + i;
      cat[r][512 + c] = (bf16)(ea * eW[c] + eB[c]);
    }
  }
  __syncthreads();

  int w = tid >> 5, lane = tid & 31;
  int hi = lane >> 4, lo = lane & 15;
  int n = w * 16 + lo;  // output column 0..255 (16 waves x 16 cols)

  // ---- f path: hidden = relu(cat @ W1f + b1f) ----
  {
    v8f acc = {};
    const bf16* bp = W1f + (size_t)n * 768 + hi * 8;
    const bf16* ap = &cat[lo][hi * 8];
    for (int kb = 0; kb < 768; kb += 32) {
      Frag a, b;
      a.u[0] = *(const u32x4*)(ap + kb);
      a.u[1] = *(const u32x4*)(ap + kb + 16);
      b.u[0] = *(const u32x4*)(bp + kb);
      b.u[1] = *(const u32x4*)(bp + kb + 16);
      acc = __builtin_amdgcn_wmma_f32_16x16x32_bf16(false, a.v, false, b.v, (short)0, acc,
                                                    false, false);
    }
    float bv = b1f[n];
    for (int i = 0; i < 8; i++) {
      int row = i + hi * 8;
      hid[row][n] = (bf16)fmaxf(acc[i] + bv, 0.f);
    }
  }
  __syncthreads();

  // ---- m = hidden @ W2f + b2f ----
  {
    v8f acc = {};
    const bf16* bp = W2f + (size_t)n * 256 + hi * 8;
    const bf16* ap = &hid[lo][hi * 8];
    for (int kb = 0; kb < 256; kb += 32) {
      Frag a, b;
      a.u[0] = *(const u32x4*)(ap + kb);
      a.u[1] = *(const u32x4*)(ap + kb + 16);
      b.u[0] = *(const u32x4*)(bp + kb);
      b.u[1] = *(const u32x4*)(bp + kb + 16);
      acc = __builtin_amdgcn_wmma_f32_16x16x32_bf16(false, a.v, false, b.v, (short)0, acc,
                                                    false, false);
    }
    float bv = b2f[n];
    for (int i = 0; i < 8; i++) mbuf[i + hi * 8][n] = acc[i] + bv;
  }
  __syncthreads();

  // ---- g path: hidden = relu(cat @ W1g + b1g) ----
  {
    v8f acc = {};
    const bf16* bp = W1g + (size_t)n * 768 + hi * 8;
    const bf16* ap = &cat[lo][hi * 8];
    for (int kb = 0; kb < 768; kb += 32) {
      Frag a, b;
      a.u[0] = *(const u32x4*)(ap + kb);
      a.u[1] = *(const u32x4*)(ap + kb + 16);
      b.u[0] = *(const u32x4*)(bp + kb);
      b.u[1] = *(const u32x4*)(bp + kb + 16);
      acc = __builtin_amdgcn_wmma_f32_16x16x32_bf16(false, a.v, false, b.v, (short)0, acc,
                                                    false, false);
    }
    float bv = b1g[n];
    for (int i = 0; i < 8; i++) {
      int row = i + hi * 8;
      hid[row][n] = (bf16)fmaxf(acc[i] + bv, 0.f);
    }
  }
  __syncthreads();

  // ---- a = hidden @ W2g + b2g; agg[dst] += m*a ----
  {
    v8f acc = {};
    const bf16* bp = W2g + (size_t)n * 256 + hi * 8;
    const bf16* ap = &hid[lo][hi * 8];
    for (int kb = 0; kb < 256; kb += 32) {
      Frag a, b;
      a.u[0] = *(const u32x4*)(ap + kb);
      a.u[1] = *(const u32x4*)(ap + kb + 16);
      b.u[0] = *(const u32x4*)(bp + kb);
      b.u[1] = *(const u32x4*)(bp + kb + 16);
      acc = __builtin_amdgcn_wmma_f32_16x16x32_bf16(false, a.v, false, b.v, (short)0, acc,
                                                    false, false);
    }
    float bv = b2g[n];
    for (int i = 0; i < 8; i++) {
      int row = i + hi * 8;
      float prod = mbuf[row][n] * (acc[i] + bv);
      atomicAdd(&agg[(size_t)dstL[row] * Hh + n], prod);
    }
  }
}

// ---------- gin = bf16([h_v | agg]) ----------
__global__ void k_gin(const float* __restrict__ hv, const float* __restrict__ agg,
                      bf16* __restrict__ gin) {
  int idx = blockIdx.x * 256 + threadIdx.x;  // Nn*512 total
  int n = idx >> 9, c = idx & 511;
  float v = (c < 256) ? hv[(size_t)n * Hh + c] : agg[(size_t)n * Hh + (c - 256)];
  gin[idx] = (bf16)v;
}

// ---------- sequential GRU scan over node dim (one persistent workgroup) ----------
__global__ __launch_bounds__(1024) void k_gru_scan(
    const float* __restrict__ gx, const float* __restrict__ Whh,
    const float* __restrict__ bhh, float* __restrict__ hv, bf16* __restrict__ hvbf) {
  __shared__ alignas(16) float h[256];
  __shared__ float gh[768];
  int tid = threadIdx.x;
  if (tid < 256) h[tid] = 0.f;
  __syncthreads();
  for (int nIdx = 0; nIdx < Nn; nIdx++) {
    if (tid < 768) {
      const f32x4* wr = (const f32x4*)(Whh + (size_t)tid * 256);
      float acc = bhh[tid];
      for (int k = 0; k < 64; k++) {
        f32x4 w4 = wr[k];
        f32x4 h4 = ((const f32x4*)h)[k];
        acc += w4.x * h4.x + w4.y * h4.y + w4.z * h4.z + w4.w * h4.w;
      }
      gh[tid] = acc;
    }
    __syncthreads();
    if (tid < 256) {
      const float* gxr = gx + (size_t)nIdx * 768;
      float r = 1.f / (1.f + __expf(-(gxr[tid] + gh[tid])));
      float z = 1.f / (1.f + __expf(-(gxr[256 + tid] + gh[256 + tid])));
      float nn = tanhf(gxr[512 + tid] + r * gh[512 + tid]);
      float hn = (1.f - z) * nn + z * h[tid];
      h[tid] = hn;
      hv[(size_t)nIdx * Hh + tid] = hn;
      hvbf[(size_t)nIdx * Hh + tid] = (bf16)hn;
    }
    __syncthreads();
  }
}

// ---------- g_emb = mean(h_v, axis=0) ----------
__global__ void k_gemb(const float* __restrict__ hv, float* __restrict__ gemb) {
  __shared__ float red[256];
  int h = blockIdx.x, tid = threadIdx.x;
  float s = 0.f;
  for (int n = tid; n < Nn; n += 256) s += hv[(size_t)n * Hh + h];
  red[tid] = s;
  __syncthreads();
  for (int st = 128; st > 0; st >>= 1) {
    if (tid < st) red[tid] += red[tid + st];
    __syncthreads();
  }
  if (tid == 0) gemb[h] = red[0] * (1.f / (float)Nn);
}

// ---------- node-prediction head: only row v_t is needed ----------
__global__ void k_np_head(const float* __restrict__ gemb, const float* __restrict__ hv,
                          const int* __restrict__ vtp,
                          const float* __restrict__ W1, const float* __restrict__ b1,
                          const float* __restrict__ W2, const float* __restrict__ b2,
                          const float* __restrict__ W3, const float* __restrict__ b3,
                          float* __restrict__ out) {
  __shared__ float nh1[256], nh2[256];
  int tid = threadIdx.x;
  int vt = *vtp;
  const float* hvt = hv + (size_t)vt * Hh;
  float acc = b1[tid];
  for (int k = 0; k < 256; k++) acc += gemb[k] * W1[k * Hh + tid];
  for (int k = 0; k < 256; k++) acc += hvt[k] * W1[(256 + k) * Hh + tid];
  nh1[tid] = fmaxf(acc, 0.f);
  __syncthreads();
  float acc2 = b2[tid];
  for (int k = 0; k < 256; k++) acc2 += nh1[k] * W2[k * Hh + tid];
  nh2[tid] = fmaxf(acc2, 0.f);
  __syncthreads();
  if (tid < 28) {
    float o = b3[tid];
    for (int k = 0; k < 256; k++) o += nh2[k] * W3[k * 28 + tid];
    out[tid] = o;
  }
}

// ---------- build alpha input rows [g_emb | h_v[v_t] | h_v[n]] bf16 ----------
__global__ void k_aalpha(const float* __restrict__ gemb, const float* __restrict__ hv,
                         const int* __restrict__ vtp, bf16* __restrict__ A) {
  int n = blockIdx.x, h = threadIdx.x;
  int vt = *vtp;
  A[(size_t)n * 768 + h]       = (bf16)gemb[h];
  A[(size_t)n * 768 + 256 + h] = (bf16)hv[(size_t)vt * Hh + h];
  A[(size_t)n * 768 + 512 + h] = (bf16)hv[(size_t)n * Hh + h];
}

// ---------- alpha tail: al = ah @ alpha_W2 + b2; alsum += al (over rows) ----------
__global__ void k_alpha_tail(const float* __restrict__ ah, const float* __restrict__ W2,
                             const float* __restrict__ b2, float* __restrict__ alsum) {
  __shared__ alignas(16) float rowv[256];
  int n = blockIdx.x, tid = threadIdx.x;  // 64 threads
  ((f32x4*)rowv)[tid] = ((const f32x4*)(ah + (size_t)n * Hh))[tid];
  __syncthreads();
  if (tid < KK) {
    float acc = b2[tid];
    for (int k = 0; k < 256; k++) acc += rowv[k] * W2[k * KK + tid];
    atomicAdd(&alsum[tid], acc);
  }
}

// ---------- lt = eh @ ep_W2 + b2 ----------
__global__ void k_ep_lt(const float* __restrict__ eh, const float* __restrict__ W2,
                        const float* __restrict__ b2, float* __restrict__ lt) {
  __shared__ alignas(8) float rowv[256];
  int n = blockIdx.x, tid = threadIdx.x;  // 128 threads
  ((f32x2*)rowv)[tid] = ((const f32x2*)(eh + (size_t)n * Hh))[tid];
  __syncthreads();
  if (tid < ET * KK) {
    float acc = b2[tid];
    for (int k = 0; k < 256; k++) acc += rowv[k] * W2[k * (ET * KK) + tid];
    lt[(size_t)n * (ET * KK) + tid] = acc;
  }
}

// ---------- p_e = sum_k alphas[k] * softmax_et(lt[n,et,k]) ----------
__global__ void k_pe(const float* __restrict__ lt, const float* __restrict__ alsum,
                     float* __restrict__ pe) {
  int n = blockIdx.x * 128 + threadIdx.x;
  float alpha[KK];
  for (int kk = 0; kk < KK; kk++) alpha[kk] = 1.f / (1.f + __expf(-alsum[kk]));
  const float* row = lt + (size_t)n * (ET * KK);
  float out[ET] = {0.f, 0.f, 0.f, 0.f};
  for (int kk = 0; kk < KK; kk++) {
    float m = row[kk];
    for (int et = 1; et < ET; et++) m = fmaxf(m, row[et * KK + kk]);
    float e[ET], den = 0.f;
    for (int et = 0; et < ET; et++) { e[et] = __expf(row[et * KK + kk] - m); den += e[et]; }
    float inv = 1.f / den;
    for (int et = 0; et < ET; et++) out[et] += alpha[kk] * e[et] * inv;
  }
  for (int et = 0; et < ET; et++) pe[(size_t)n * ET + et] = out[et];
}

// ============================= host orchestration =============================
extern "C" void kernel_launch(void* const* d_in, const int* in_sizes, int n_in,
                              void* d_out, int out_size, void* d_ws, size_t ws_size,
                              hipStream_t stream) {
  (void)in_sizes; (void)n_in; (void)out_size; (void)ws_size;
  const float* x      = (const float*)d_in[0];
  const int*   eidx   = (const int*)d_in[1];
  const float* eattr  = (const float*)d_in[2];
  const int*   vtp    = (const int*)d_in[3];
  const float* node_W = (const float*)d_in[4];
  const float* node_b = (const float*)d_in[5];
  const float* edge_W = (const float*)d_in[6];
  const float* edge_b = (const float*)d_in[7];
  const float* fW1 = (const float*)d_in[8];
  const float* fb1 = (const float*)d_in[9];
  const float* fW2 = (const float*)d_in[10];
  const float* fb2 = (const float*)d_in[11];
  const float* gW1 = (const float*)d_in[12];
  const float* gb1 = (const float*)d_in[13];
  const float* gW2 = (const float*)d_in[14];
  const float* gb2 = (const float*)d_in[15];
  const float* Wih = (const float*)d_in[16];
  const float* Whh = (const float*)d_in[17];
  const float* bih = (const float*)d_in[18];
  const float* bhh = (const float*)d_in[19];
  const float* aW1 = (const float*)d_in[20];
  const float* ab1 = (const float*)d_in[21];
  const float* aW2 = (const float*)d_in[22];
  const float* ab2 = (const float*)d_in[23];
  const float* npW1 = (const float*)d_in[24];
  const float* npb1 = (const float*)d_in[25];
  const float* npW2 = (const float*)d_in[26];
  const float* npb2 = (const float*)d_in[27];
  const float* npW3 = (const float*)d_in[28];
  const float* npb3 = (const float*)d_in[29];
  const float* epW1 = (const float*)d_in[30];
  const float* epb1 = (const float*)d_in[31];
  const float* epW2 = (const float*)d_in[32];
  const float* epb2 = (const float*)d_in[33];

  char* ws = (char*)d_ws;
  size_t off = 0;
  auto take = [&](size_t bytes) -> char* {
    char* p = ws + off;
    off = (off + bytes + 255) & ~(size_t)255;
    return p;
  };
  float* hv     = (float*)take((size_t)Nn * Hh * 4);
  bf16*  hvbf   = (bf16*)take((size_t)Nn * Hh * 2);
  float* agg    = (float*)take((size_t)Nn * Hh * 4);
  bf16*  ginbf  = (bf16*)take((size_t)Nn * 512 * 2);
  float* gx     = (float*)take((size_t)Nn * 768 * 4);
  bf16*  wf1t   = (bf16*)take((size_t)LL * 768 * 256 * 2);
  bf16*  wf2t   = (bf16*)take((size_t)LL * 256 * 256 * 2);
  bf16*  wg1t   = (bf16*)take((size_t)LL * 768 * 256 * 2);
  bf16*  wg2t   = (bf16*)take((size_t)LL * 256 * 256 * 2);
  bf16*  wihbf  = (bf16*)take((size_t)LL * 768 * 512 * 2);
  bf16*  wa1t   = (bf16*)take((size_t)768 * 256 * 2);
  bf16*  wep1t  = (bf16*)take((size_t)256 * 256 * 2);
  float* gemb   = (float*)take(256 * 4);
  bf16*  Aalpha = (bf16*)take((size_t)Nn * 768 * 2);
  float* ah     = (float*)take((size_t)Nn * Hh * 4);
  float* eh     = (float*)take((size_t)Nn * Hh * 4);
  float* lt     = (float*)take((size_t)Nn * ET * KK * 4);
  float* alsum  = (float*)take(32 * 4);

  // ---- weight prep (bf16, B transposed to [Nout,K]; Wih already [Nout,K]) ----
  k_transcvt<<<1920, 256, 0, stream>>>(fW1, wf1t, 768, 256, LL * 768 * 256);
  k_transcvt<<<1280, 256, 0, stream>>>(fW2, wf2t, 256, 256, LL * 256 * 256);
  k_transcvt<<<1920, 256, 0, stream>>>(gW1, wg1t, 768, 256, LL * 768 * 256);
  k_transcvt<<<1280, 256, 0, stream>>>(gW2, wg2t, 256, 256, LL * 256 * 256);
  k_convert<<<1920, 256, 0, stream>>>(Wih, wihbf, LL * 768 * 512);
  k_transcvt<<<768, 256, 0, stream>>>(aW1, wa1t, 768, 256, 768 * 256);
  k_transcvt<<<256, 256, 0, stream>>>(epW1, wep1t, 256, 256, 256 * 256);

  // ---- initial node / edge embeddings ----
  k_node_proj<<<Nn, 256, 0, stream>>>(x, node_W, node_b, hv, hvbf);

  // ---- L message-passing layers ----
  for (int l = 0; l < LL; l++) {
    hipMemsetAsync(agg, 0, (size_t)Nn * Hh * 4, stream);
    k_edge_mlp<<<Ee / 16, 512, 0, stream>>>(
        hvbf, eidx, eattr, edge_W, edge_b,
        wf1t + (size_t)l * 768 * 256, fb1 + l * 256,
        wf2t + (size_t)l * 256 * 256, fb2 + l * 256,
        wg1t + (size_t)l * 768 * 256, gb1 + l * 256,
        wg2t + (size_t)l * 256 * 256, gb2 + l * 256, agg);
    k_gin<<<Nn * 512 / 256, 256, 0, stream>>>(hv, agg, ginbf);
    k_gemm_bf16<<<(Nn / 16) * (768 / 16) / 8, 256, 0, stream>>>(
        ginbf, wihbf + (size_t)l * 768 * 512, bih + l * 768, gx, Nn, 768, 512, 0);
    k_gru_scan<<<1, 1024, 0, stream>>>(gx, Whh + (size_t)l * 768 * 256, bhh + l * 768,
                                       hv, hvbf);
  }

  // ---- heads ----
  float* out = (float*)d_out;
  k_gemb<<<256, 256, 0, stream>>>(hv, gemb);
  k_np_head<<<1, 256, 0, stream>>>(gemb, hv, vtp, npW1, npb1, npW2, npb2, npW3, npb3, out);
  k_aalpha<<<Nn, 256, 0, stream>>>(gemb, hv, vtp, Aalpha);
  k_gemm_bf16<<<(Nn / 16) * (256 / 16) / 8, 256, 0, stream>>>(Aalpha, wa1t, ab1, ah,
                                                              Nn, 256, 768, 1);
  hipMemsetAsync(alsum, 0, 32 * 4, stream);
  k_alpha_tail<<<Nn, 64, 0, stream>>>(ah, aW2, ab2, alsum);
  k_gemm_bf16<<<(Nn / 16) * (256 / 16) / 8, 256, 0, stream>>>(hvbf, wep1t, epb1, eh,
                                                              Nn, 256, 256, 1);
  k_ep_lt<<<Nn, 128, 0, stream>>>(eh, epW2, epb2, lt);
  k_pe<<<Nn / 128, 128, 0, stream>>>(lt, alsum, out + 28);
  hipMemcpyAsync(out + 28 + Nn * ET, hv, (size_t)Nn * Hh * 4,
                 hipMemcpyDeviceToDevice, stream);
}
